// MluQuantLinearInt8_25881472926614
// MI455X (gfx1250) — 8-bit matrix operand order repaired and checked by reading, compiled, not run on hardware
//
#include <hip/hip_runtime.h>

typedef __attribute__((ext_vector_type(8))) int v8i;

#define S_DIM 4096
#define K_DIM 4096
#define N_DIM 16384
#define QMAXF 127.0f

#define BM 128
#define BN 128
#define BK 64
#define NIT (K_DIM / BK)
#define LDSS 80

#if defined(__HIP_DEVICE_COMPILE__) &&                                   \
    __has_builtin(__builtin_amdgcn_global_load_async_to_lds_b128) &&     \
    __has_builtin(__builtin_amdgcn_s_wait_asynccnt)
#define USE_ASYNC_LDS 1
typedef int v4i128 __attribute__((vector_size(16)));
typedef __attribute__((address_space(1))) v4i128 gvec128;
typedef __attribute__((address_space(3))) v4i128 lvec128;

__device__ __forceinline__ void async_cp16(const signed char* g, signed char* l)
{
    __builtin_amdgcn_global_load_async_to_lds_b128(
        (gvec128*)g, (lvec128*)l, 0, 0);
}
#else
#define USE_ASYNC_LDS 0
#endif

__global__ __launch_bounds__(256) void pack_weight_int8(
    const int* __restrict__ wi32, int* __restrict__ wpacked, long total4)
{
    long i = (long)blockIdx.x * blockDim.x + threadIdx.x;
    if (i < total4) {
        int4 v = ((const int4*)wi32)[i];
        wpacked[i] = (v.x & 255) | ((v.y & 255) << 8) |
                     ((v.z & 255) << 16) | ((v.w & 255) << 24);
    }
}

__global__ __launch_bounds__(256) void quant_rows(
    const float* __restrict__ x, signed char* __restrict__ q,
    float* __restrict__ xscale)
{
    const int row = blockIdx.x;
    const int t = threadIdx.x;
    const float* xr = x + (size_t)row * K_DIM;

    float amax = 0.0f;
    for (int i = t; i < K_DIM; i += 256)
        amax = fmaxf(amax, fabsf(xr[i]));

    for (int off = 16; off > 0; off >>= 1)
        amax = fmaxf(amax, __shfl_xor(amax, off, 32));

    __shared__ float red[8];
    if ((t & 31) == 0) red[t >> 5] = amax;
    __syncthreads();
    if (t < 8) {
        float v = red[t];
        for (int off = 4; off > 0; off >>= 1)
            v = fmaxf(v, __shfl_xor(v, off, 32));
        if (t == 0) red[0] = v;
    }
    __syncthreads();
    amax = red[0];

    const float denom = fmaxf(amax, 1e-8f);
    const float scale = denom / QMAXF;
    const float rcp   = QMAXF / denom;
    if (t == 0) xscale[row] = scale;

    int* qo = (int*)(q + (size_t)row * K_DIM);
    for (int c = 0; c < 4; ++c) {
        int idx = c * 1024 + t * 4;
        float4 v = *(const float4*)(xr + idx);
        int r0 = (int)fminf(fmaxf(rintf(v.x * rcp), -QMAXF), QMAXF);
        int r1 = (int)fminf(fmaxf(rintf(v.y * rcp), -QMAXF), QMAXF);
        int r2 = (int)fminf(fmaxf(rintf(v.z * rcp), -QMAXF), QMAXF);
        int r3 = (int)fminf(fmaxf(rintf(v.w * rcp), -QMAXF), QMAXF);
        qo[idx >> 2] = (r0 & 255) | ((r1 & 255) << 8) |
                       ((r2 & 255) << 16) | ((r3 & 255) << 24);
    }
}

__global__ __launch_bounds__(256) void gemm_iu8(
    const signed char* __restrict__ qa,
    const signed char* __restrict__ wb,
    const float* __restrict__ xscale,
    const float* __restrict__ wscale,
    float* __restrict__ out)
{
    __shared__ __align__(16) signed char As[2][BM * LDSS];
    __shared__ __align__(16) signed char Bs[2][BN * LDSS];

    const int t    = threadIdx.x;
    const int lane = t & 31;
    const int wave = t >> 5;
    const int wm   = wave >> 2;
    const int wn   = wave & 3;

    const int m0 = blockIdx.y * BM;
    const int n0 = blockIdx.x * BN;

    const int srow = t >> 1;
    const int scol = (t & 1) * 32;
    const int soff = srow * LDSS + scol;

    const signed char* aptr = qa + (size_t)(m0 + srow) * K_DIM + scol;
    const signed char* bptr = wb + (size_t)(n0 + srow) * K_DIM + scol;

    const int a_hi = (lane >= 16) ? 16 : 0;
    const int b_hi = (lane >= 16) ? 16 : 0;
    const int sub  = lane & 15;

    const int aoff = (wm * 64 + sub) * LDSS + a_hi;
    const int boff = (wn * 32 + sub) * LDSS + b_hi;

    v8i acc[4][2] = {};

    auto fetch = [&](int pp) {
#if USE_ASYNC_LDS
        async_cp16(aptr,      &As[pp][soff]);
        async_cp16(aptr + 16, &As[pp][soff + 16]);
        async_cp16(bptr,      &Bs[pp][soff]);
        async_cp16(bptr + 16, &Bs[pp][soff + 16]);
#else
        int4 av0 = *(const int4*)(aptr);
        int4 av1 = *(const int4*)(aptr + 16);
        int4 bv0 = *(const int4*)(bptr);
        int4 bv1 = *(const int4*)(bptr + 16);
        *(int4*)(&As[pp][soff])      = av0;
        *(int4*)(&As[pp][soff + 16]) = av1;
        *(int4*)(&Bs[pp][soff])      = bv0;
        *(int4*)(&Bs[pp][soff + 16]) = bv1;
#endif
        aptr += BK; bptr += BK;
    };

    auto sync = [&]() {
#if USE_ASYNC_LDS
        __builtin_amdgcn_s_wait_asynccnt(0);
#endif
        __syncthreads();
    };

    auto compute = [&](int pp) {
        const signed char* Ab = As[pp];
        const signed char* Bb = Bs[pp];

        v8i bfrag[2];
        #pragma unroll
        for (int j = 0; j < 2; ++j) {
            const signed char* bb = Bb + boff + j * 16 * LDSS;
            int4 lo = *(const int4*)(bb);
            int4 hi = *(const int4*)(bb + 32);
            v8i f;
            f[0] = lo.x; f[1] = lo.y; f[2] = lo.z; f[3] = lo.w;
            f[4] = hi.x; f[5] = hi.y; f[6] = hi.z; f[7] = hi.w;
            bfrag[j] = f;
        }

        #pragma unroll
        for (int i = 0; i < 4; ++i) {
            const signed char* ab = Ab + aoff + i * 16 * LDSS;
            int2 p0 = *(const int2*)(ab);
            int2 p1 = *(const int2*)(ab + 8);
            int2 p2 = *(const int2*)(ab + 32);
            int2 p3 = *(const int2*)(ab + 40);
            v8i af;
            af[0] = p0.x; af[1] = p0.y; af[2] = p1.x; af[3] = p1.y;
            af[4] = p2.x; af[5] = p2.y; af[6] = p3.x; af[7] = p3.y;

            #pragma unroll
            for (int j = 0; j < 2; ++j) {
                acc[i][j] = __builtin_amdgcn_wmma_i32_16x16x64_iu8(
                    true, af, true, bfrag[j], acc[i][j], false, false);
            }
        }
    };

    fetch(0);
    sync();

    #pragma unroll 1
    for (int it = 0; it < NIT - 2; it += 2) {
        fetch(1); compute(0); sync();
        fetch(0); compute(1); sync();
    }
    fetch(1); compute(0); sync();
    compute(1);

    const int d_hi = (lane >= 16) ? 8 : 0;
    #pragma unroll
    for (int j = 0; j < 2; ++j) {
        const int n = n0 + wn * 32 + j * 16 + sub;
        const float wsj = wscale[n];
        #pragma unroll
        for (int i = 0; i < 4; ++i) {
            const int mbase = m0 + wm * 64 + i * 16 + d_hi;
            #pragma unroll
            for (int r = 0; r < 8; ++r) {
                const int m = mbase + r;
                out[(size_t)m * N_DIM + n] =
                    (float)acc[i][j][r] * xscale[m] * wsj;
            }
        }
    }
}

extern "C" void kernel_launch(void* const* d_in, const int* in_sizes, int n_in,
                              void* d_out, int out_size, void* d_ws, size_t ws_size,
                              hipStream_t stream) {
    const float* x      = (const float*)d_in[0];
    const int*   w_i32  = (const int*)d_in[1];
    const float* wscale = (const float*)d_in[2];
    float*       out    = (float*)d_out;

    char* ws = (char*)d_ws;
    float*       xscale  = (float*)ws;
    signed char* q       = (signed char*)(ws + 16384);
    signed char* wpacked = (signed char*)(ws + 16384 +
                                          (size_t)S_DIM * K_DIM);

    {
        long total4 = (long)N_DIM * K_DIM / 4;
        int threads = 256;
        long blocks = (total4 + threads - 1) / threads;
        pack_weight_int8<<<(unsigned)blocks, threads, 0, stream>>>(
            w_i32, (int*)wpacked, total4);
    }

    quant_rows<<<S_DIM, 256, 0, stream>>>(x, q, xscale);

    {
        dim3 grid(N_DIM / BN, S_DIM / BM);
        gemm_iu8<<<grid, 256, 0, stream>>>(q, wpacked, xscale, wscale, out);
    }
}
